// Attention_90984587198810
// MI455X (gfx1250) — matrix operand order repaired and checked by reading, not compiled, not run on hardware
//
#include <hip/hip_runtime.h>
#include <hip/hip_bf16.h>

#define B_    2
#define S_    2048
#define D_    4096
#define H_    32
#define KV_   8
#define HD_   128
#define NREP_ 4

typedef __attribute__((ext_vector_type(16))) __bf16 v16bf;
typedef __attribute__((ext_vector_type(8)))  __bf16 v8bf;
typedef __attribute__((ext_vector_type(8)))  float  v8f;

#define WMMA_BF16(a, b, c) \
  __builtin_amdgcn_wmma_f32_16x16x32_bf16(false, (a), false, (b), (short)0, (c), false, false)

// ---- fragment loaders (contiguous k order k = 16*(lane/16) + elem, same in A and B) ------------------------------
// A (16x32, row-major source): lane holds one row; elems 0..15 at K=khi+0..15
// (khi = 0 for lanes 0-15, 16 for lanes 16-31).
__device__ __forceinline__ v16bf load_a_frag(const __bf16* p) {
  union { v16bf v; v8bf h[2]; } u;
  u.h[0] = *reinterpret_cast<const v8bf*>(p);
  u.h[1] = *reinterpret_cast<const v8bf*>(p + 8);
  return u.v;
}
// B (32x16): lane n holds column n over 16 contiguous K values -> one 32B load.
__device__ __forceinline__ v16bf load_b_frag(const __bf16* p) {
  return *reinterpret_cast<const v16bf*>(p);
}

__device__ __forceinline__ float rowmax16(float t) {
  t = fmaxf(t, __shfl_xor(t, 1, 16));
  t = fmaxf(t, __shfl_xor(t, 2, 16));
  t = fmaxf(t, __shfl_xor(t, 4, 16));
  t = fmaxf(t, __shfl_xor(t, 8, 16));
  return t;
}
__device__ __forceinline__ float rowsum16(float t) {
  t += __shfl_xor(t, 1, 16);
  t += __shfl_xor(t, 2, 16);
  t += __shfl_xor(t, 4, 16);
  t += __shfl_xor(t, 8, 16);
  return t;
}

// ---- elementwise prep kernels -------------------------------------------
__global__ void cvt_f32_to_bf16(const float* __restrict__ in, __bf16* __restrict__ out, long n) {
  long t = (long)blockIdx.x * blockDim.x + threadIdx.x;
  if (t < n) out[t] = (__bf16)in[t];
}

// in: f32 (K x N) row-major -> out: bf16 (N x K) row-major (B^T for WMMA B-frags)
__global__ void wtrans_bf16(const float* __restrict__ in, __bf16* __restrict__ out,
                            int K, int N, long total) {
  long t = (long)blockIdx.x * blockDim.x + threadIdx.x;
  if (t >= total) return;
  int  k = (int)(t % K);
  long n = t / K;
  out[t] = (__bf16)in[(long)k * N + n];
}

// RoPE + (B,S,nh,HD) -> (B,nh,S,HD) transpose, bf16 in/out, math in f32
__global__ void rope_kernel(const __bf16* __restrict__ in, const float* __restrict__ cs,
                            const float* __restrict__ sn, __bf16* __restrict__ out,
                            int nh, long total_pairs) {
  long t = (long)blockIdx.x * blockDim.x + threadIdx.x;
  if (t >= total_pairs) return;
  int  i = (int)(t % (HD_ / 2)); long r = t / (HD_ / 2);
  int  h = (int)(r % nh);  r /= nh;
  int  s = (int)(r % S_);  int b = (int)(r / S_);
  const __bf16* p = in + (((long)(b * S_ + s) * nh + h) * HD_) + 2 * i;
  float e = (float)p[0], o = (float)p[1];
  float c = cs[s * (HD_ / 2) + i], sv = sn[s * (HD_ / 2) + i];
  __bf16* q = out + (((long)(b * nh + h) * S_ + s) * HD_) + 2 * i;
  q[0] = (__bf16)(e * c - o * sv);
  q[1] = (__bf16)(e * sv + o * c);
}

// (B,S,KV,HD) -> (B,KV,HD,S)  (V^T so V B-fragments are K-contiguous)
__global__ void vtrans_kernel(const __bf16* __restrict__ in, __bf16* __restrict__ out, long total) {
  long t = (long)blockIdx.x * blockDim.x + threadIdx.x;
  if (t >= total) return;
  int d = (int)(t % HD_); long r = t / HD_;
  int g = (int)(r % KV_); r /= KV_;
  int s = (int)(r % S_);  int b = (int)(r / S_);
  out[(((long)(b * KV_ + g) * HD_ + d) * S_) + s] = in[t];
}

// ---- bf16 GEMM: C[M,N] = A[M,K] * Bt[N,K]^T, f32 accumulate -------------
// block = 128 thr = 4 waves; block tile 128x128; wave tile 64x64 (16 WMMA/K-step)
__global__ __launch_bounds__(128) void gemm_bf16_kernel(
    const __bf16* __restrict__ A, const __bf16* __restrict__ Bt,
    __bf16* __restrict__ Cb, float* __restrict__ Cf, int N, int K) {
  const int lane = threadIdx.x & 31;
  const int wave = threadIdx.x >> 5;
  const int m0 = blockIdx.y * 128 + (wave >> 1) * 64;
  const int n0 = blockIdx.x * 128 + (wave & 1) * 64;
  const int row   = lane & 15;
  const int khiA  = (lane & 16) ? 16 : 0;
  const int koffB = (lane & 16) ? 16 : 0;

  const __bf16* ap[4];
  const __bf16* bp[4];
#pragma unroll
  for (int i = 0; i < 4; ++i) ap[i] = A  + (long)(m0 + 16 * i + row) * K + khiA;
#pragma unroll
  for (int j = 0; j < 4; ++j) bp[j] = Bt + (long)(n0 + 16 * j + row) * K + koffB;

  v8f acc[4][4] = {};
  for (int k = 0; k < K; k += 32) {
    v16bf a[4], b[4];
#pragma unroll
    for (int i = 0; i < 4; ++i) a[i] = load_a_frag(ap[i] + k);
#pragma unroll
    for (int j = 0; j < 4; ++j) b[j] = load_b_frag(bp[j] + k);
#pragma unroll
    for (int i = 0; i < 4; ++i)
#pragma unroll
      for (int j = 0; j < 4; ++j)
        acc[i][j] = WMMA_BF16(a[i], b[j], acc[i][j]);
  }
  const int crow = (lane & 16) ? 8 : 0;
#pragma unroll
  for (int i = 0; i < 4; ++i)
#pragma unroll
    for (int j = 0; j < 4; ++j)
#pragma unroll
      for (int r = 0; r < 8; ++r) {
        const long idx = (long)(m0 + 16 * i + crow + r) * N + (n0 + 16 * j + row);
        if (Cf) Cf[idx] = acc[i][j][r];
        else    Cb[idx] = (__bf16)acc[i][j][r];
      }
}

// ---- flash attention: 1 wave per (b, h, 16 query rows) ------------------
__global__ __launch_bounds__(32) void flash_attn_kernel(
    const __bf16* __restrict__ Q,   // (B,H,S,HD)
    const __bf16* __restrict__ Km,  // (B,KV,S,HD)
    const __bf16* __restrict__ Vt,  // (B,KV,HD,S)
    __bf16* __restrict__ O) {       // (B,S,H*HD)
  __shared__ __align__(32) __bf16 Pl[16][32];
  const int lane = threadIdx.x;
  const int qt = blockIdx.x, h = blockIdx.y, b = blockIdx.z;
  const int g  = h / NREP_;
  const int q0 = qt * 16;
  const int row   = lane & 15;
  const int half  = lane >> 4;
  const int khiA  = half * 16;
  const int koffB = half * 16;

  const __bf16* qb = Q  + (((long)b * H_  + h) * S_ + q0) * HD_;
  const __bf16* kb = Km + ((long)b * KV_ + g) * (long)S_ * HD_;
  const __bf16* vb = Vt + ((long)b * KV_ + g) * (long)HD_ * S_;

  v16bf qf[4];
#pragma unroll
  for (int kd = 0; kd < 4; ++kd)
    qf[kd] = load_a_frag(qb + row * HD_ + kd * 32 + khiA);

  v8f o[8] = {};
  float m[8], l[8];
#pragma unroll
  for (int r = 0; r < 8; ++r) { m[r] = -1e30f; l[r] = 0.f; }

  const float scale = 0.08838834764831845f;  // 1/sqrt(128)
  const int nsteps = (q0 + 16 + 31) >> 5;    // causal: keys < q0+16
  for (int kt = 0; kt < nsteps; ++kt) {
    const int j0 = kt * 32;
    v8f s0 = {}, s1 = {};
#pragma unroll
    for (int kd = 0; kd < 4; ++kd) {
      v16bf b0 = load_b_frag(kb + (long)(j0 + row)      * HD_ + kd * 32 + koffB);
      v16bf b1 = load_b_frag(kb + (long)(j0 + 16 + row) * HD_ + kd * 32 + koffB);
      s0 = WMMA_BF16(qf[kd], b0, s0);
      s1 = WMMA_BF16(qf[kd], b1, s1);
    }
    float corr[8];
#pragma unroll
    for (int r = 0; r < 8; ++r) {
      const int qrow = q0 + half * 8 + r;
      float a0 = s0[r] * scale;
      float a1 = s1[r] * scale;
      if (j0 + row      > qrow) a0 = -1e30f;   // causal mask
      if (j0 + 16 + row > qrow) a1 = -1e30f;
      float mn = fmaxf(m[r], rowmax16(fmaxf(a0, a1)));
      float p0 = __expf(a0 - mn);
      float p1 = __expf(a1 - mn);
      corr[r] = __expf(m[r] - mn);
      m[r] = mn;
      l[r] = l[r] * corr[r] + rowsum16(p0 + p1);
      s0[r] = p0; s1[r] = p1;
    }
#pragma unroll
    for (int nc = 0; nc < 8; ++nc)
#pragma unroll
      for (int r = 0; r < 8; ++r) o[nc][r] *= corr[r];

    // stage P (C-layout -> A-layout) through LDS; same-wave DS ops are in order
#pragma unroll
    for (int r = 0; r < 8; ++r) {
      Pl[half * 8 + r][row]      = (__bf16)s0[r];
      Pl[half * 8 + r][row + 16] = (__bf16)s1[r];
    }
    asm volatile("s_wait_dscnt 0" ::: "memory");
    union { v16bf v; v8bf h2[2]; } pu;
    pu.h2[0] = *reinterpret_cast<const v8bf*>(&Pl[row][khiA]);
    pu.h2[1] = *reinterpret_cast<const v8bf*>(&Pl[row][khiA + 8]);
#pragma unroll
    for (int nc = 0; nc < 8; ++nc) {
      v16bf vf = load_b_frag(vb + (long)(nc * 16 + row) * S_ + j0 + koffB);
      o[nc] = WMMA_BF16(pu.v, vf, o[nc]);
    }
  }
#pragma unroll
  for (int nc = 0; nc < 8; ++nc)
#pragma unroll
    for (int r = 0; r < 8; ++r) {
      const int qrow = q0 + half * 8 + r;
      float val = o[nc][r] / l[r];
      O[(long)(b * S_ + qrow) * (H_ * HD_) + h * HD_ + nc * 16 + row] = (__bf16)val;
    }
}

// ---- orchestration ------------------------------------------------------
extern "C" void kernel_launch(void* const* d_in, const int* in_sizes, int n_in,
                              void* d_out, int out_size, void* d_ws, size_t ws_size,
                              hipStream_t stream) {
  const float* x  = (const float*)d_in[0];
  const float* fc = (const float*)d_in[1];
  const float* fs = (const float*)d_in[2];
  // d_in[3] = mask (unused: causal mask applied analytically)
  const float* wq = (const float*)d_in[4];
  const float* wk = (const float*)d_in[5];
  const float* wv = (const float*)d_in[6];
  const float* wo = (const float*)d_in[7];
  // d_in[8] = start_pos (== 0)
  float* out = (float*)d_out;

  const long BS = (long)B_ * S_;           // 4096
  char*  ws  = (char*)d_ws;
  size_t off = 0;
  auto alloc = [&](size_t bytes) -> void* {
    void* p = ws + off;
    off = (off + bytes + 255) & ~(size_t)255;
    return p;
  };
  __bf16* xb   = (__bf16*)alloc(BS * D_ * 2);
  __bf16* wqT  = (__bf16*)alloc((size_t)(H_ * HD_) * D_ * 2);
  __bf16* wkT  = (__bf16*)alloc((size_t)(KV_ * HD_) * D_ * 2);
  __bf16* wvT  = (__bf16*)alloc((size_t)(KV_ * HD_) * D_ * 2);
  __bf16* woT  = (__bf16*)alloc((size_t)D_ * (H_ * HD_) * 2);
  __bf16* qpre = (__bf16*)alloc(BS * H_ * HD_ * 2);
  __bf16* kpre = (__bf16*)alloc(BS * KV_ * HD_ * 2);
  __bf16* vpre = (__bf16*)alloc(BS * KV_ * HD_ * 2);
  __bf16* qr   = (__bf16*)alloc(BS * H_ * HD_ * 2);
  __bf16* kr   = (__bf16*)alloc(BS * KV_ * HD_ * 2);
  __bf16* vT   = (__bf16*)alloc(BS * KV_ * HD_ * 2);
  __bf16* attn = qpre;  // qpre is dead after rope_q; reuse for attention output
  if (off > ws_size) return;

  const int TB = 256;
  // 1) convert x and transpose weights to bf16
  {
    long n = BS * D_;
    cvt_f32_to_bf16<<<(n + TB - 1) / TB, TB, 0, stream>>>(x, xb, n);
  }
  {
    long n = (long)D_ * (H_ * HD_);
    wtrans_bf16<<<(n + TB - 1) / TB, TB, 0, stream>>>(wq, wqT, D_, H_ * HD_, n);
  }
  {
    long n = (long)D_ * (KV_ * HD_);
    wtrans_bf16<<<(n + TB - 1) / TB, TB, 0, stream>>>(wk, wkT, D_, KV_ * HD_, n);
    wtrans_bf16<<<(n + TB - 1) / TB, TB, 0, stream>>>(wv, wvT, D_, KV_ * HD_, n);
  }
  {
    long n = (long)(H_ * HD_) * D_;
    wtrans_bf16<<<(n + TB - 1) / TB, TB, 0, stream>>>(wo, woT, H_ * HD_, D_, n);
  }
  // 2) QKV projections
  {
    dim3 g(4096 / 128, (unsigned)(BS / 128));
    gemm_bf16_kernel<<<g, 128, 0, stream>>>(xb, wqT, qpre, nullptr, H_ * HD_, D_);
  }
  {
    dim3 g(1024 / 128, (unsigned)(BS / 128));
    gemm_bf16_kernel<<<g, 128, 0, stream>>>(xb, wkT, kpre, nullptr, KV_ * HD_, D_);
    gemm_bf16_kernel<<<g, 128, 0, stream>>>(xb, wvT, vpre, nullptr, KV_ * HD_, D_);
  }
  // 3) RoPE (+ head-major transpose) and V transpose
  {
    long n = BS * H_ * (HD_ / 2);
    rope_kernel<<<(n + TB - 1) / TB, TB, 0, stream>>>(qpre, fc, fs, qr, H_, n);
  }
  {
    long n = BS * KV_ * (HD_ / 2);
    rope_kernel<<<(n + TB - 1) / TB, TB, 0, stream>>>(kpre, fc, fs, kr, KV_, n);
  }
  {
    long n = BS * KV_ * HD_;
    vtrans_kernel<<<(n + TB - 1) / TB, TB, 0, stream>>>(vpre, vT, n);
  }
  // 4) causal flash attention
  {
    dim3 g(S_ / 16, H_, B_);
    flash_attn_kernel<<<g, 32, 0, stream>>>(qr, kr, vT, attn);
  }
  // 5) output projection -> f32 out
  {
    dim3 g(4096 / 128, (unsigned)(BS / 128));
    gemm_bf16_kernel<<<g, 128, 0, stream>>>(attn, woT, nullptr, out, D_, H_ * HD_);
  }
}
